// MultiScaleSSM_81011673137626
// MI455X (gfx1250) — compile-verified, matrix operand order checked by reading, not run on hardware
//
#include <hip/hip_runtime.h>

typedef unsigned int uint;
typedef unsigned short ushort;
typedef __attribute__((ext_vector_type(16))) __bf16 v16bf;
typedef __attribute__((ext_vector_type(8)))  float  v8f;

#define BATCH   4
#define SEQ     4096
#define DMODEL  1024
#define DINNER  2048
#define NSTATE  16
#define MTOT    (BATCH * SEQ)   // 16384

// ---------------- helpers ----------------

__device__ __forceinline__ uint f2bf_bits(float f) {
    uint u = __float_as_uint(f);
    return (u + 0x7FFFu + ((u >> 16) & 1u)) >> 16;   // round-to-nearest-even
}
__device__ __forceinline__ uint pack_bf2(float lo, float hi) {
    return f2bf_bits(lo) | (f2bf_bits(hi) << 16);
}

union FragBF { uint u[8]; v16bf v; };

// async copy: 16 bytes global -> LDS, per lane; tracked by ASYNCcnt
__device__ __forceinline__ void async_g2l_b128(uint lds_byte_off, const void* gaddr) {
    asm volatile("global_load_async_to_lds_b128 %0, %1, off"
                 :: "v"(lds_byte_off), "v"(gaddr) : "memory");
}

// ---------------- f32 -> bf16 conversion (with zero padding) ----------------
// 8 elements per thread; n_src and n_dst multiples of 8.

__global__ __launch_bounds__(256)
void cvt_bf16_kernel(const float* __restrict__ src, ushort* __restrict__ dst,
                     int n_src, int n_dst) {
    int i8 = (blockIdx.x * 256 + threadIdx.x) * 8;
    if (i8 >= n_dst) return;
    uint4 o;
    if (i8 < n_src) {
        float4 p0 = *(const float4*)(src + i8);
        float4 p1 = *(const float4*)(src + i8 + 4);
        o.x = pack_bf2(p0.x, p0.y);
        o.y = pack_bf2(p0.z, p0.w);
        o.z = pack_bf2(p1.x, p1.y);
        o.w = pack_bf2(p1.z, p1.w);
    } else {
        o = make_uint4(0u, 0u, 0u, 0u);   // zero padding rows (W_B / W_C)
    }
    *(uint4*)(dst + i8) = o;
}

// ---------------- WMMA GEMM: C[M,N] = A[M,K](bf16) * W[N,K](bf16)^T ----------------
// ACT: 0 = none, 1 = softplus(x + bias), 2 = x * sigmoid(gate[0])
// WB : also write bf16 copy of result columns gn < N2 to C2 (ld = N2)
// Block: 256 threads (8 waves). Block tile 64(M) x 128(N), K staged 32 at a time.
// Wave w: rows (w&1)*32, cols (w>>1)*32 -> 2x2 grid of 16x16 wmma tiles.
// Requires: M % 64 == 0, K % 32 == 0, W padded to N % 128 == 0 rows.
// Both tiles staged with global_load_async_to_lds_b128, double-buffered.

#define TPITCH 40   // ushorts = 80 B rows: 16B-aligned -> ds b128; 20-dword stride
                    // is conflict-free within each 16-lane half

template<int ACT, bool WB>
__global__ __launch_bounds__(256)
void wmma_gemm_kernel(const ushort* __restrict__ A, int lda,
                      const ushort* __restrict__ W, int ldw,
                      float* __restrict__ C, int ldo,
                      ushort* __restrict__ C2, int N2,
                      int M, int N, int K,
                      const float* __restrict__ bias,
                      const float* __restrict__ gate) {
    __shared__ ushort As[2][64  * TPITCH];
    __shared__ ushort Bs[2][128 * TPITCH];

    const int tid  = threadIdx.x;
    const int lane = tid & 31;
    const int wid  = tid >> 5;
    const int half = lane >> 4;      // 0: lanes 0-15, 1: lanes 16-31
    const int row  = lane & 15;

    const int m0 = (wid & 1) * 32;   // wave row offset in block tile (0/32)
    const int n0 = (wid >> 1) * 32;  // wave col offset in block tile (0..96)

    const int blkM = blockIdx.y * 64;
    const int blkN = blockIdx.x * 128;

    // staging: rows of 64 B (32 bf16); one 16B chunk per thread for A (64 rows),
    // two for W (128 rows)
    const int sr  = tid >> 2;                    // 0..63
    const int sc  = (tid & 3) * 16;              // byte offset in row
    const ushort* aSrc  = A + (size_t)(blkM + sr) * lda + (sc >> 1);
    const ushort* wSrc0 = W + (size_t)(blkN + sr) * ldw + (sc >> 1);
    const ushort* wSrc1 = W + (size_t)(blkN + 64 + sr) * ldw + (sc >> 1);
    const uint aDst  = (uint)(size_t)&As[0][sr * TPITCH] + (uint)sc;
    const uint wDst0 = (uint)(size_t)&Bs[0][sr * TPITCH] + (uint)sc;
    const uint wDst1 = (uint)(size_t)&Bs[0][(64 + sr) * TPITCH] + (uint)sc;
    const uint aStride = (uint)(64  * TPITCH * sizeof(ushort));
    const uint bStride = (uint)(128 * TPITCH * sizeof(ushort));

    v8f acc00 = {}, acc01 = {}, acc10 = {}, acc11 = {};

    // prefetch first tiles into buffer 0
    async_g2l_b128(aDst,  aSrc);
    async_g2l_b128(wDst0, wSrc0);
    async_g2l_b128(wDst1, wSrc1);

    int buf = 0;
    for (int kc = 0; kc < K; kc += 32) {
        __syncthreads();   // all waves done gathering from buf^1 -> reusable

        const bool pref = (kc + 32 < K);
        if (pref) {        // async prefetch of next tiles into other buffer
            const int nb = buf ^ 1;
            async_g2l_b128(aDst  + nb * aStride, aSrc  + kc + 32);
            async_g2l_b128(wDst0 + nb * bStride, wSrc0 + kc + 32);
            async_g2l_b128(wDst1 + nb * bStride, wSrc1 + kc + 32);
        }

        // current tiles complete (async loads complete in order); the just-issued
        // 3 prefetch copies may remain outstanding and overlap the WMMAs below
        if (pref) asm volatile("s_wait_asynccnt 0x3" ::: "memory");
        else      asm volatile("s_wait_asynccnt 0x0" ::: "memory");
        __syncthreads();

        // fragments per ISA 16-bit layout: lane holds M/N = row, halves split K;
        // VGPR v: v<4 -> k = 2v + 8*half ; v>=4 -> k = 16 + 2(v-4) + 8*half
        // -> u[0..3] and u[4..7] are 16B-aligned dword quads (ds_load_b128)
        FragBF a0, a1, b0, b1;
        #pragma unroll
        for (int v = 0; v < 8; ++v) {
            const int kb = ((v < 4) ? (v * 2) : (16 + (v - 4) * 2)) + half * 8;
            a0.u[v] = *(const uint*)&As[buf][(m0 + row)      * TPITCH + kb];
            a1.u[v] = *(const uint*)&As[buf][(m0 + 16 + row) * TPITCH + kb];
            b0.u[v] = *(const uint*)&Bs[buf][(n0 + row)      * TPITCH + kb];
            b1.u[v] = *(const uint*)&Bs[buf][(n0 + 16 + row) * TPITCH + kb];
        }
        acc00 = __builtin_amdgcn_wmma_f32_16x16x32_bf16(false, a0.v, false, b0.v,
                                                        (short)0, acc00, false, false);
        acc01 = __builtin_amdgcn_wmma_f32_16x16x32_bf16(false, a0.v, false, b1.v,
                                                        (short)0, acc01, false, false);
        acc10 = __builtin_amdgcn_wmma_f32_16x16x32_bf16(false, a1.v, false, b0.v,
                                                        (short)0, acc10, false, false);
        acc11 = __builtin_amdgcn_wmma_f32_16x16x32_bf16(false, a1.v, false, b1.v,
                                                        (short)0, acc11, false, false);
        buf ^= 1;
    }

    // epilogue: C/D layout -> VGPR r holds M = r + 8*half, N = lane&15
    float gscale = 1.0f;
    if (ACT == 2) gscale = 1.0f / (1.0f + __expf(-gate[0]));

    #pragma unroll
    for (int tm = 0; tm < 2; ++tm) {
        #pragma unroll
        for (int tn = 0; tn < 2; ++tn) {
            v8f acc = tm ? (tn ? acc11 : acc10) : (tn ? acc01 : acc00);
            const int gn = blkN + n0 + tn * 16 + row;
            if (gn >= N) continue;
            const int gmb = blkM + m0 + tm * 16 + half * 8;
            float bv = 0.0f;
            if (ACT == 1) bv = bias[gn];
            float* dst = C + (size_t)gmb * ldo + gn;
            #pragma unroll
            for (int r = 0; r < 8; ++r) {
                float v = acc[r];
                if (ACT == 1) {
                    v += bv;
                    v = fmaxf(v, 0.0f) + log1pf(__expf(-fabsf(v)));  // softplus
                } else if (ACT == 2) {
                    v *= gscale;
                }
                dst[(size_t)r * ldo] = v;
                if (WB && gn < N2)
                    C2[(size_t)(gmb + r) * N2 + gn] = (ushort)f2bf_bits(v);
            }
        }
    }
}

// ---------------- selective scan ----------------
// grid: (DINNER/256, BATCH); each thread owns one channel d with h[16] in regs.
// Reads f32 dt/x/z, writes bf16 gated y (A operand of the output projection).
// B/C vectors double-buffered in LDS -> one barrier per timestep.

__global__ __launch_bounds__(256)
void scan_kernel(const float* __restrict__ A_log,
                 const float* __restrict__ xz,      // (B*L, 2*DINNER) f32
                 const float* __restrict__ dtbuf,   // (B*L, DINNER)   f32
                 const float* __restrict__ Bbuf,    // (B*L, NSTATE)   f32
                 const float* __restrict__ Cbuf,    // (B*L, NSTATE)   f32
                 const float* __restrict__ Dvec,
                 ushort*      __restrict__ ybf) {   // (B*L, DINNER)   bf16 out
    __shared__ float sB[2][NSTATE];
    __shared__ float sC[2][NSTATE];

    const int b = blockIdx.y;
    const int d = blockIdx.x * 256 + threadIdx.x;
    const size_t tok0 = (size_t)b * SEQ;

    float a[NSTATE], h[NSTATE];
    #pragma unroll
    for (int n = 0; n < NSTATE; ++n) {
        a[n] = -__expf(A_log[d * NSTATE + n]);
        h[n] = 0.0f;
    }
    const float Dd = Dvec[d];

    // preload t = 0
    if (threadIdx.x < NSTATE)
        sB[0][threadIdx.x] = Bbuf[tok0 * NSTATE + threadIdx.x];
    else if (threadIdx.x < 2 * NSTATE)
        sC[0][threadIdx.x - NSTATE] = Cbuf[tok0 * NSTATE + (threadIdx.x - NSTATE)];
    __syncthreads();

    for (int t = 0; t < SEQ; ++t) {
        const size_t tok = tok0 + t;
        const int cur = t & 1, nxt = cur ^ 1;

        if (t + 1 < SEQ) {   // prefetch next step's B/C into the other slot
            if (threadIdx.x < NSTATE)
                sB[nxt][threadIdx.x] = Bbuf[(tok + 1) * NSTATE + threadIdx.x];
            else if (threadIdx.x < 2 * NSTATE)
                sC[nxt][threadIdx.x - NSTATE] = Cbuf[(tok + 1) * NSTATE + (threadIdx.x - NSTATE)];
        }

        const float dt = dtbuf[tok * DINNER + d];
        const float xv = xz[tok * (2 * DINNER) + d];
        const float zv = xz[tok * (2 * DINNER) + DINNER + d];

        float y = 0.0f;
        #pragma unroll
        for (int n = 0; n < NSTATE; ++n) {
            float ab = __expf(fminf(dt * a[n], 20.0f));
            h[n] = ab * h[n] + (dt * sB[cur][n]) * xv;
            y += h[n] * sC[cur][n];
        }
        y += xv * Dd;
        y *= zv / (1.0f + __expf(-zv));          // * silu(z)
        ybf[tok * DINNER + d] = (ushort)f2bf_bits(y);
        __syncthreads();
    }
}

// ---------------- launch ----------------

extern "C" void kernel_launch(void* const* d_in, const int* in_sizes, int n_in,
                              void* d_out, int out_size, void* d_ws, size_t ws_size,
                              hipStream_t stream) {
    (void)in_sizes; (void)n_in; (void)out_size; (void)ws_size;

    const float* x     = (const float*)d_in[0];  // (B,L,1024)
    const float* W_in  = (const float*)d_in[1];  // (4096,1024)
    const float* A_log = (const float*)d_in[2];  // (2048,16)
    const float* W_B   = (const float*)d_in[3];  // (16,2048)
    const float* W_C   = (const float*)d_in[4];  // (16,2048)
    const float* W_dt  = (const float*)d_in[5];  // (2048,2048)
    const float* b_dt  = (const float*)d_in[6];  // (2048,)
    const float* Dvec  = (const float*)d_in[7];  // (2048,)
    const float* gate  = (const float*)d_in[8];  // (1,)
    const float* W_out = (const float*)d_in[9];  // (1024,2048)
    float* out = (float*)d_out;

    // workspace layout: f32 region, then bf16 region
    float* xz  = (float*)d_ws;                       // MTOT x 4096 (x_ssm | z)
    float* dtb = xz  + (size_t)MTOT * (2 * DINNER);  // MTOT x 2048 dt
    float* Bb  = dtb + (size_t)MTOT * DINNER;        // MTOT x 16
    float* Cb  = Bb  + (size_t)MTOT * NSTATE;        // MTOT x 16
    ushort* xb  = (ushort*)(Cb + (size_t)MTOT * NSTATE);  // MTOT x 1024 (x bf16)
    ushort* xsb = xb  + (size_t)MTOT * DMODEL;            // MTOT x 2048 (x_ssm bf16)
    ushort* yb  = xsb + (size_t)MTOT * DINNER;            // MTOT x 2048 (y bf16)
    ushort* Wib = yb  + (size_t)MTOT * DINNER;            // 4096 x 1024
    ushort* Wdb = Wib + (size_t)(2 * DINNER) * DMODEL;    // 2048 x 2048
    ushort* Wob = Wdb + (size_t)DINNER * DINNER;          // 1024 x 2048
    ushort* WBb = Wob + (size_t)DMODEL * DINNER;          // 128  x 2048 (padded)
    ushort* WCb = WBb + (size_t)128 * DINNER;             // 128  x 2048 (padded)

    dim3 blk(256);

    // 0) one-time f32 -> bf16 conversions (weights + x; W_B/W_C zero-padded)
    {
        int n;
        n = MTOT * DMODEL;
        cvt_bf16_kernel<<<n / (256 * 8), blk, 0, stream>>>(x, xb, n, n);
        n = 2 * DINNER * DMODEL;
        cvt_bf16_kernel<<<n / (256 * 8), blk, 0, stream>>>(W_in, Wib, n, n);
        n = DINNER * DINNER;
        cvt_bf16_kernel<<<n / (256 * 8), blk, 0, stream>>>(W_dt, Wdb, n, n);
        n = DMODEL * DINNER;
        cvt_bf16_kernel<<<n / (256 * 8), blk, 0, stream>>>(W_out, Wob, n, n);
        int nd = 128 * DINNER, ns = NSTATE * DINNER;
        cvt_bf16_kernel<<<nd / (256 * 8), blk, 0, stream>>>(W_B, WBb, ns, nd);
        cvt_bf16_kernel<<<nd / (256 * 8), blk, 0, stream>>>(W_C, WCb, ns, nd);
    }

    // 1) xz = x @ W_in^T  (f32 out + bf16 copy of x_ssm cols) (N=4096, K=1024)
    wmma_gemm_kernel<0, true><<<dim3(2 * DINNER / 128, MTOT / 64), blk, 0, stream>>>(
        xb, DMODEL, Wib, DMODEL, xz, 2 * DINNER, xsb, DINNER,
        MTOT, 2 * DINNER, DMODEL, nullptr, nullptr);

    // 2) dt = softplus(x_ssm @ W_dt^T + b_dt)   (N=2048, K=2048)
    wmma_gemm_kernel<1, false><<<dim3(DINNER / 128, MTOT / 64), blk, 0, stream>>>(
        xsb, DINNER, Wdb, DINNER, dtb, DINNER, nullptr, 0,
        MTOT, DINNER, DINNER, b_dt, nullptr);

    // 3) B_mat = x_ssm @ W_B^T    (N=16, padded weight rows)
    wmma_gemm_kernel<0, false><<<dim3(1, MTOT / 64), blk, 0, stream>>>(
        xsb, DINNER, WBb, DINNER, Bb, NSTATE, nullptr, 0,
        MTOT, NSTATE, DINNER, nullptr, nullptr);

    // 4) C_mat = x_ssm @ W_C^T    (N=16, padded weight rows)
    wmma_gemm_kernel<0, false><<<dim3(1, MTOT / 64), blk, 0, stream>>>(
        xsb, DINNER, WCb, DINNER, Cb, NSTATE, nullptr, 0,
        MTOT, NSTATE, DINNER, nullptr, nullptr);

    // 5) selective scan -> bf16 y
    scan_kernel<<<dim3(DINNER / 256, BATCH), blk, 0, stream>>>(
        A_log, xz, dtb, Bb, Cb, Dvec, yb);

    // 6) out = (y @ W_out^T) * sigmoid(gate)    (N=1024, K=2048)
    wmma_gemm_kernel<2, false><<<dim3(DMODEL / 128, MTOT / 64), blk, 0, stream>>>(
        yb, DINNER, Wob, DINNER, out, DMODEL, nullptr, 0,
        MTOT, DMODEL, DINNER, nullptr, gate);
}
